// net_point_84524956385828
// MI455X (gfx1250) — matrix operand order repaired and checked by reading, not compiled, not run on hardware
//
#include <hip/hip_runtime.h>
#include <math.h>

typedef __attribute__((ext_vector_type(16))) _Float16 v16h;
typedef __attribute__((ext_vector_type(8)))  _Float16 v8h;
typedef __attribute__((ext_vector_type(8)))  float    v8f;
typedef __attribute__((ext_vector_type(4)))  float    f4;

#define N_NODES 100000
#define N_EDGES 1600000

// ---- f16 transposed-weight region (half-element offsets inside d_ws) ----
#define H_WM1  0        // [512][128]
#define H_WM2  65536    // [64][512]
#define H_WM3  98304    // [64][64]
#define H_PARS 102400   // [64][64]   relu(2*P0)^T
#define H_WI1  106496   // [128][128]
#define H_WI2  122880   // [128][128]
#define H_WP1  139264   // [128][128]
#define H_WL   155648   // [128][160] (K padded 131->160 with zeros)
#define H_WG   176128   // [128][128]
#define H_WF   192512   // [64][128]
#define H_END  200704

// ---- byte offsets of activation buffers inside d_ws ----
#define B_LOGITS 401408ull                 // f16 [N][64]
#define B_Q      13201408ull               // f16 [N][64]
#define B_XHA    26001408ull               // f16 [N][128]
#define B_XHB    51601408ull               // f16 [N][128]
#define B_POS    77201408ull               // f32 [N][3]
#define B_EWRAW  78401408ull               // f32 [E]
#define B_AGG    84801408ull               // u32 [N][128] (order-encoded)
#define B_ACC    136001408ull              // f64 [2]  sum, sumsq
#define B_PRM    136001424ull              // f32 [2]  mean, scale

#define ENC_NEG_INF 0x007FFFFFu

__device__ __forceinline__ int laneid() { return (int)(threadIdx.x & 31u); }

// A fragment (16x32 f16, row-major source with leading dim ld, tile already offset)
__device__ __forceinline__ v16h a_frag(const _Float16* A, int ld, int k) {
  int l = laneid();
  int r = l & 15;
  int hi = l >> 4;
  const _Float16* p = A + (size_t)r * ld + k + hi * 16;
  v8h x0 = *(const v8h*)p;
  v8h x1 = *(const v8h*)(p + 8);
  v16h o;
#pragma unroll
  for (int i = 0; i < 8; ++i) { o[i] = x0[i]; o[i + 8] = x1[i]; }
  return o;
}

// B fragment (32x16 f16) from transposed weights WT[out][in], leading dim ld
__device__ __forceinline__ v16h b_frag(const _Float16* WT, int ld, int col0, int k) {
  int l = laneid();
  int c = col0 + (l & 15);
  int hi = l >> 4;
  return *(const v16h*)(WT + (size_t)c * ld + k + hi * 16);
}

__device__ __forceinline__ v8f wmma16(v16h a, v16h b, v8f c) {
  return __builtin_amdgcn_wmma_f32_16x16x32_f16(false, a, false, b, (short)0, c, false, false);
}

__device__ __forceinline__ v8f bias_acc(const float* b, int col0) {
  float bv = b[col0 + (laneid() & 15)];
  v8f c;
#pragma unroll
  for (int i = 0; i < 8; ++i) c[i] = bv;
  return c;
}

__device__ __forceinline__ v8f zero_acc() {
  v8f c;
#pragma unroll
  for (int i = 0; i < 8; ++i) c[i] = 0.f;
  return c;
}

// Store D tile (16x16 f32 acc) to f16 row-major [ld], optional relu
__device__ __forceinline__ void store_d_h(_Float16* D, int ld, int col0, v8f c, bool relu) {
  int l = laneid();
  int hi = l >> 4;
  int col = col0 + (l & 15);
#pragma unroll
  for (int v = 0; v < 8; ++v) {
    float x = c[v];
    if (relu && x < 0.f) x = 0.f;
    D[(size_t)(v + 8 * hi) * ld + col] = (_Float16)x;
  }
}

// =============================== prep weights ===============================
__global__ void prep_weights(const float* __restrict__ Wm1, const float* __restrict__ Wm2,
                             const float* __restrict__ Wm3, const float* __restrict__ P0,
                             const float* __restrict__ Wi1, const float* __restrict__ Wi2,
                             const float* __restrict__ Wp1, const float* __restrict__ Wl,
                             const float* __restrict__ Wg, const float* __restrict__ Wf,
                             _Float16* __restrict__ wh) {
  int t = blockIdx.x * blockDim.x + threadIdx.x;
  if (t < 65536) {                       // Wm1^T: [512][128], src (128,512)
    int o = t >> 7, i = t & 127;
    wh[H_WM1 + t] = (_Float16)Wm1[i * 512 + o];
  } else if (t < 98304) {                // Wm2^T: [64][512], src (512,64)
    int u = t - 65536; int o = u >> 9, i = u & 511;
    wh[H_WM2 + u] = (_Float16)Wm2[i * 64 + o];
  } else if (t < 102400) {               // Wm3^T: [64][64], src (64,64)
    int u = t - 98304; int o = u >> 6, i = u & 63;
    wh[H_WM3 + u] = (_Float16)Wm3[i * 64 + o];
  } else if (t < 106496) {               // parsing^T = relu(2*P0)^T
    int u = t - 102400; int o = u >> 6, i = u & 63;
    float v = 2.0f * P0[i * 64 + o];
    wh[H_PARS + u] = (_Float16)(v > 0.f ? v : 0.f);
  } else if (t < 122880) {               // Wi1^T [128][128]
    int u = t - 106496; int o = u >> 7, i = u & 127;
    wh[H_WI1 + u] = (_Float16)Wi1[i * 128 + o];
  } else if (t < 139264) {               // Wi2^T
    int u = t - 122880; int o = u >> 7, i = u & 127;
    wh[H_WI2 + u] = (_Float16)Wi2[i * 128 + o];
  } else if (t < 155648) {               // Wp1^T
    int u = t - 139264; int o = u >> 7, i = u & 127;
    wh[H_WP1 + u] = (_Float16)Wp1[i * 128 + o];
  } else if (t < 176128) {               // Wl^T [128][160], src (131,128), K pad
    int u = t - 155648; int o = u / 160, i = u % 160;
    wh[H_WL + u] = (i < 131) ? (_Float16)Wl[i * 128 + o] : (_Float16)0.f;
  } else if (t < 192512) {               // Wg^T
    int u = t - 176128; int o = u >> 7, i = u & 127;
    wh[H_WG + u] = (_Float16)Wg[i * 128 + o];
  } else if (t < H_END) {                // Wf^T [64][128], src (128,64)
    int u = t - 192512; int o = u >> 7, i = u & 127;
    wh[H_WF + u] = (_Float16)Wf[i * 64 + o];
  }
}

// =============================== fused node MLP =============================
// 2 waves per block, one 16-row tile per wave. Exact grid: N/32 blocks.
__global__ void __launch_bounds__(64)
node_mlp(const float* __restrict__ x, const _Float16* __restrict__ wh,
         const float* __restrict__ bm1, const float* __restrict__ bm2,
         const float* __restrict__ bm3, const float* __restrict__ bi1,
         const float* __restrict__ bi2, const float* __restrict__ bp1,
         const float* __restrict__ Wp2, const float* __restrict__ bp2,
         _Float16* __restrict__ logits_h, _Float16* __restrict__ q_h,
         _Float16* __restrict__ xh_h, float* __restrict__ pos) {
  __shared__ _Float16 s_h1[2][16 * 512];  // 32 KB
  __shared__ _Float16 s_bx[2][16 * 128];  // 8 KB
  __shared__ _Float16 s_h2[2][16 * 64];   // 4 KB
  __shared__ _Float16 s_lg[2][16 * 64];   // 4 KB

  const int w = threadIdx.x >> 5;
  const int l = laneid();
  const int row0 = (blockIdx.x * 2 + w) * 16;
  _Float16* h1 = s_h1[w];
  _Float16* bx = s_bx[w];
  _Float16* h2 = s_h2[w];
  _Float16* lg = s_lg[w];

  // load x tile -> f16 LDS (16x128)
  for (int i = l; i < 512; i += 32) {
    int r = i >> 5, c4 = i & 31;
    f4 v = ((const f4*)x)[(size_t)(row0 + r) * 32 + c4];
    int o = r * 128 + c4 * 4;
    bx[o] = (_Float16)v.x; bx[o + 1] = (_Float16)v.y;
    bx[o + 2] = (_Float16)v.z; bx[o + 3] = (_Float16)v.w;
  }
  __syncthreads();

  // S1: h1 = relu(x @ Wm1 + bm1)  [16x512]
  for (int ct = 0; ct < 32; ++ct) {
    v8f c = bias_acc(bm1, ct * 16);
#pragma unroll
    for (int ks = 0; ks < 4; ++ks)
      c = wmma16(a_frag(bx, 128, ks * 32), b_frag(wh + H_WM1, 128, ct * 16, ks * 32), c);
    store_d_h(h1, 512, ct * 16, c, true);
  }
  __syncthreads();

  // S2: h2 = relu(h1 @ Wm2 + bm2)  [16x64]
  for (int ct = 0; ct < 4; ++ct) {
    v8f c = bias_acc(bm2, ct * 16);
#pragma unroll
    for (int ks = 0; ks < 16; ++ks)
      c = wmma16(a_frag(h1, 512, ks * 32), b_frag(wh + H_WM2, 512, ct * 16, ks * 32), c);
    store_d_h(h2, 64, ct * 16, c, true);
  }
  __syncthreads();

  // S3: logits = h2 @ Wm3 + bm3  [16x64] -> LDS + global f16
  for (int ct = 0; ct < 4; ++ct) {
    v8f c = bias_acc(bm3, ct * 16);
#pragma unroll
    for (int ks = 0; ks < 2; ++ks)
      c = wmma16(a_frag(h2, 64, ks * 32), b_frag(wh + H_WM3, 64, ct * 16, ks * 32), c);
    store_d_h(lg, 64, ct * 16, c, false);
    store_d_h(logits_h + (size_t)row0 * 64, 64, ct * 16, c, false);
  }
  __syncthreads();

  // S4: q = logits @ parsing  -> global f16
  for (int ct = 0; ct < 4; ++ct) {
    v8f c = zero_acc();
#pragma unroll
    for (int ks = 0; ks < 2; ++ks)
      c = wmma16(a_frag(lg, 64, ks * 32), b_frag(wh + H_PARS, 64, ct * 16, ks * 32), c);
    store_d_h(q_h + (size_t)row0 * 64, 64, ct * 16, c, false);
  }

  // S5: t1 = relu(x @ Wi1 + bi1)  -> reuse h1 as [16x128]
  for (int ct = 0; ct < 8; ++ct) {
    v8f c = bias_acc(bi1, ct * 16);
#pragma unroll
    for (int ks = 0; ks < 4; ++ks)
      c = wmma16(a_frag(bx, 128, ks * 32), b_frag(wh + H_WI1, 128, ct * 16, ks * 32), c);
    store_d_h(h1, 128, ct * 16, c, true);
  }
  __syncthreads();

  // S6: xh = t1 @ Wi2 + bi2 -> global f16 + LDS (overwrite bx)
  for (int ct = 0; ct < 8; ++ct) {
    v8f c = bias_acc(bi2, ct * 16);
#pragma unroll
    for (int ks = 0; ks < 4; ++ks)
      c = wmma16(a_frag(h1, 128, ks * 32), b_frag(wh + H_WI2, 128, ct * 16, ks * 32), c);
    store_d_h(xh_h + (size_t)row0 * 128, 128, ct * 16, c, false);
    store_d_h(bx, 128, ct * 16, c, false);
  }
  __syncthreads();

  // S7: p1 = relu(xh @ Wp1 + bp1) -> h1
  for (int ct = 0; ct < 8; ++ct) {
    v8f c = bias_acc(bp1, ct * 16);
#pragma unroll
    for (int ks = 0; ks < 4; ++ks)
      c = wmma16(a_frag(bx, 128, ks * 32), b_frag(wh + H_WP1, 128, ct * 16, ks * 32), c);
    store_d_h(h1, 128, ct * 16, c, true);
  }
  __syncthreads();

  // S8: pos = p1 @ Wp2 + bp2  (128 -> 3, VALU)
  if (l < 16) {
    int r = l;
    for (int oc = 0; oc < 3; ++oc) {
      float s = bp2[oc];
      for (int k = 0; k < 128; ++k)
        s += (float)h1[r * 128 + k] * Wp2[k * 3 + oc];
      pos[(size_t)(row0 + r) * 3 + oc] = s;
    }
  }
}

// =============================== edge weights ===============================
__global__ void __launch_bounds__(256)
edge_dot(const int* __restrict__ ei, const _Float16* __restrict__ lg,
         const _Float16* __restrict__ q, float* __restrict__ ew_raw,
         double* __restrict__ acc) {
  int e = blockIdx.x * blockDim.x + threadIdx.x;
  int s = ei[e], d = ei[N_EDGES + e];
  const v8h* ps = (const v8h*)(lg + (size_t)s * 64);
  const v8h* pt = (const v8h*)(q + (size_t)d * 64);
  float sum = 0.f;
#pragma unroll
  for (int j = 0; j < 8; ++j) {
    v8h a = ps[j], b = pt[j];
#pragma unroll
    for (int k = 0; k < 8; ++k) sum += (float)a[k] * (float)b[k];
  }
  ew_raw[e] = sum;
  float ss = sum * sum;
  float rs = sum;
  for (int o = 16; o; o >>= 1) { rs += __shfl_xor(rs, o, 32); ss += __shfl_xor(ss, o, 32); }
  if ((threadIdx.x & 31u) == 0) {
    atomicAdd(acc, (double)rs);
    atomicAdd(acc + 1, (double)ss);
  }
}

__global__ void ew_finalize(const double* __restrict__ acc, float* __restrict__ prm) {
  double S = acc[0], SS = acc[1];
  double n = (double)N_EDGES;
  double mean = S / n;
  double var = (SS - S * S / n) / (n - 1.0);
  prm[0] = (float)mean;
  prm[1] = (float)sqrt(1e-4 / var);
}

// =============================== aggregation ================================
__global__ void __launch_bounds__(256)
agg_init(unsigned* __restrict__ agg) {
  size_t i = (size_t)blockIdx.x * blockDim.x + threadIdx.x;
  agg[i] = ENC_NEG_INF;
}

// 2 waves/block, 32 edges per wave (two M-tiles share each B fragment).
// Exact grid: E/64 blocks.
__global__ void __launch_bounds__(64)
edge_msg(const int* __restrict__ ei, const _Float16* __restrict__ xh,
         const float* __restrict__ pos, const float* __restrict__ ew_raw,
         const float* __restrict__ prm, const _Float16* __restrict__ wh,
         const float* __restrict__ bl, unsigned* __restrict__ agg) {
  __shared__ _Float16 s_feat[2][32 * 160];  // 10 KB/wave
  __shared__ float s_ew[2][32];
  __shared__ int s_dst[2][32];

  const int w = threadIdx.x >> 5;
  const int l = laneid();
  const int e0 = (blockIdx.x * 2 + w) * 32;
  _Float16* feat = s_feat[w];

  const float mean = prm[0], scale = prm[1];

  // gather: one lane per edge (32 edges)
  {
    int e = e0 + l;
    int sidx = ei[e], didx = ei[N_EDGES + e];
    const v8h* px = (const v8h*)(xh + (size_t)sidx * 128);
    v8h* pf = (v8h*)(feat + l * 160);
#pragma unroll
    for (int j = 0; j < 16; ++j) pf[j] = px[j];
#pragma unroll
    for (int oc = 0; oc < 3; ++oc)
      feat[l * 160 + 128 + oc] =
          (_Float16)(pos[(size_t)sidx * 3 + oc] - pos[(size_t)didx * 3 + oc]);
    for (int c = 131; c < 160; ++c) feat[l * 160 + c] = (_Float16)0.f;
    s_ew[w][l] = (ew_raw[e] - mean) * scale + 1.0f;
    s_dst[w][l] = didx;
  }
  __syncthreads();

  const int hi = l >> 4;
  for (int ct = 0; ct < 8; ++ct) {
    v8f c0 = bias_acc(bl, ct * 16);
    v8f c1 = c0;
#pragma unroll
    for (int ks = 0; ks < 5; ++ks) {
      v16h b = b_frag(wh + H_WL, 160, ct * 16, ks * 32);
      c0 = wmma16(a_frag(feat, 160, ks * 32), b, c0);
      c1 = wmma16(a_frag(feat + 16 * 160, 160, ks * 32), b, c1);
    }
    int col = ct * 16 + (l & 15);
#pragma unroll
    for (int v = 0; v < 8; ++v) {
      int M = v + 8 * hi;
      {
        float val = c0[v] * s_ew[w][M];
        unsigned u = __float_as_uint(val);
        unsigned enc = (u & 0x80000000u) ? ~u : (u | 0x80000000u);
        atomicMax(&agg[(size_t)s_dst[w][M] * 128 + col], enc);
      }
      {
        float val = c1[v] * s_ew[w][16 + M];
        unsigned u = __float_as_uint(val);
        unsigned enc = (u & 0x80000000u) ? ~u : (u | 0x80000000u);
        atomicMax(&agg[(size_t)s_dst[w][16 + M] * 128 + col], enc);
      }
    }
  }
}

// decode agg + xh = relu(agg @ Wg + bg). 1 wave/block, 32 rows (two M-tiles
// share each B fragment). Exact grid: N/32.
__global__ void __launch_bounds__(32)
node_update(const unsigned* __restrict__ agg, const _Float16* __restrict__ wh,
            const float* __restrict__ bg, _Float16* __restrict__ xh_out) {
  __shared__ _Float16 s_a[32 * 128];  // 8 KB
  const int l = laneid();
  const int row0 = blockIdx.x * 32;

  for (int i = l; i < 4096; i += 32) {
    unsigned e = agg[(size_t)row0 * 128 + i];
    float f;
    if (e == ENC_NEG_INF) f = 0.f;                    // empty segment -> 0
    else if (e & 0x80000000u) f = __uint_as_float(e ^ 0x80000000u);
    else f = __uint_as_float(~e);
    s_a[i] = (_Float16)f;
  }
  __syncthreads();

  for (int ct = 0; ct < 8; ++ct) {
    v8f c0 = bias_acc(bg, ct * 16);
    v8f c1 = c0;
#pragma unroll
    for (int ks = 0; ks < 4; ++ks) {
      v16h b = b_frag(wh + H_WG, 128, ct * 16, ks * 32);
      c0 = wmma16(a_frag(s_a, 128, ks * 32), b, c0);
      c1 = wmma16(a_frag(s_a + 16 * 128, 128, ks * 32), b, c1);
    }
    store_d_h(xh_out + (size_t)row0 * 128, 128, ct * 16, c0, true);
    store_d_h(xh_out + (size_t)(row0 + 16) * 128, 128, ct * 16, c1, true);
  }
}

// out = xh @ Wf + bf. 1 wave/block, 32 rows, A fragments straight from global
// f16, B shared across both M-tiles. Exact grid: N/32.
__global__ void __launch_bounds__(32)
final_proj(const _Float16* __restrict__ xh, const _Float16* __restrict__ wh,
           const float* __restrict__ bf, float* __restrict__ out) {
  const int l = laneid();
  const int row0 = blockIdx.x * 32;
  const int hi = l >> 4;
  const _Float16* A0 = xh + (size_t)row0 * 128;
  const _Float16* A1 = xh + (size_t)(row0 + 16) * 128;

  for (int ct = 0; ct < 4; ++ct) {
    v8f c0 = bias_acc(bf, ct * 16);
    v8f c1 = c0;
#pragma unroll
    for (int ks = 0; ks < 4; ++ks) {
      v16h b = b_frag(wh + H_WF, 128, ct * 16, ks * 32);
      c0 = wmma16(a_frag(A0, 128, ks * 32), b, c0);
      c1 = wmma16(a_frag(A1, 128, ks * 32), b, c1);
    }
    int col = ct * 16 + (l & 15);
#pragma unroll
    for (int v = 0; v < 8; ++v) {
      out[(size_t)(row0 + v + 8 * hi) * 64 + col] = c0[v];
      out[(size_t)(row0 + 16 + v + 8 * hi) * 64 + col] = c1[v];
    }
  }
}

// ================================ launcher ==================================
extern "C" void kernel_launch(void* const* d_in, const int* in_sizes, int n_in,
                              void* d_out, int out_size, void* d_ws, size_t ws_size,
                              hipStream_t stream) {
  const float* x   = (const float*)d_in[0];
  const int*   ei  = (const int*)d_in[1];
  const float* Wm1 = (const float*)d_in[2];  const float* bm1 = (const float*)d_in[3];
  const float* Wm2 = (const float*)d_in[4];  const float* bm2 = (const float*)d_in[5];
  const float* Wm3 = (const float*)d_in[6];  const float* bm3 = (const float*)d_in[7];
  const float* P0  = (const float*)d_in[8];
  const float* Wi1 = (const float*)d_in[9];  const float* bi1 = (const float*)d_in[10];
  const float* Wi2 = (const float*)d_in[11]; const float* bi2 = (const float*)d_in[12];
  const float* Wp1 = (const float*)d_in[13]; const float* bp1 = (const float*)d_in[14];
  const float* Wp2 = (const float*)d_in[15]; const float* bp2 = (const float*)d_in[16];
  const float* Wl  = (const float*)d_in[17]; const float* bl  = (const float*)d_in[18];
  const float* Wg  = (const float*)d_in[19]; const float* bg  = (const float*)d_in[20];
  const float* Wf  = (const float*)d_in[21]; const float* bf  = (const float*)d_in[22];

  char* ws = (char*)d_ws;
  _Float16* wh     = (_Float16*)ws;
  _Float16* lg_h   = (_Float16*)(ws + B_LOGITS);
  _Float16* q_h    = (_Float16*)(ws + B_Q);
  _Float16* xhA    = (_Float16*)(ws + B_XHA);
  _Float16* xhB    = (_Float16*)(ws + B_XHB);
  float*    pos    = (float*)(ws + B_POS);
  float*    ew_raw = (float*)(ws + B_EWRAW);
  unsigned* agg    = (unsigned*)(ws + B_AGG);
  double*   acc    = (double*)(ws + B_ACC);
  float*    prm    = (float*)(ws + B_PRM);

  // 1. weights -> f16 transposed
  prep_weights<<<H_END / 256, 256, 0, stream>>>(Wm1, Wm2, Wm3, P0, Wi1, Wi2, Wp1,
                                                Wl, Wg, Wf, wh);
  // 2. fused node MLPs (logits, q, xh0, pos)
  node_mlp<<<N_NODES / 32, 64, 0, stream>>>(x, wh, bm1, bm2, bm3, bi1, bi2, bp1,
                                            Wp2, bp2, lg_h, q_h, xhA, pos);
  // 3. edge weights: dot + mean/var
  hipMemsetAsync(acc, 0, 2 * sizeof(double), stream);
  edge_dot<<<N_EDGES / 256, 256, 0, stream>>>(ei, lg_h, q_h, ew_raw, acc);
  ew_finalize<<<1, 1, 0, stream>>>(acc, prm);

  // 4. two PointNetConv layers
  _Float16* cur = xhA;
  _Float16* nxt = xhB;
  for (int layer = 0; layer < 2; ++layer) {
    agg_init<<<(N_NODES * 128) / 256, 256, 0, stream>>>(agg);
    edge_msg<<<N_EDGES / 64, 64, 0, stream>>>(ei, cur, pos, ew_raw, prm, wh, bl, agg);
    node_update<<<N_NODES / 32, 32, 0, stream>>>(agg, wh, bg, nxt);
    _Float16* t = cur; cur = nxt; nxt = t;
  }

  // 5. final projection to d_out
  final_proj<<<N_NODES / 32, 32, 0, stream>>>(cur, wh, bf, (float*)d_out);
}
